// GIN_dc_34067680592270
// MI455X (gfx1250) — compile-verified, matrix operand order checked by reading, not run on hardware
//
#include <hip/hip_runtime.h>
#include <hip/hip_bf16.h>

typedef __attribute__((ext_vector_type(16))) _Float16 v16h;
typedef __attribute__((ext_vector_type(8)))  _Float16 v8h;
typedef __attribute__((ext_vector_type(8)))  float    v8f;

#define GN_N 50000
#define GN_E 800000
#define GN_G 500

// ---------------------------------------------------------------- utilities
__global__ void k_fill(float* __restrict__ p, int n, float v) {
    int t = blockIdx.x * blockDim.x + threadIdx.x;
    if (t < n) p[t] = v;
}

__global__ void k_deg(const int* __restrict__ dst, float* __restrict__ deg, int E) {
    int t = blockIdx.x * blockDim.x + threadIdx.x;
    if (t < E) atomicAdd(&deg[dst[t]], 1.0f);
}

__global__ void k_dinv(const float* __restrict__ deg, float* __restrict__ dinv, int n) {
    int t = blockIdx.x * blockDim.x + threadIdx.x;
    if (t < n) dinv[t] = rsqrtf(deg[t]);
}

// f16 conversion (optionally fused add): dst[r,ldd] = (half)(s1[r,lds] + s2[r,lds])
__global__ void k_cvt(_Float16* __restrict__ dst, const float* __restrict__ s1,
                      const float* __restrict__ s2, int rows, int cols,
                      int lds, int ldd) {
    int t = blockIdx.x * blockDim.x + threadIdx.x;
    if (t >= rows * cols) return;
    int r = t / cols, cidx = t - r * cols;
    float v = s1[(long long)r * lds + cidx];
    if (s2) v += s2[(long long)r * lds + cidx];
    dst[(long long)r * ldd + cidx] = (_Float16)v;
}

// ------------------------------------------------------------- WMMA GEMM
// out[N,64] = act( Ah(f16) @ W[K,64] + bias ), K in {16,64,128} (compile-time).
// One wave per 16x16 tile; 8 waves/block = 2 row-tiles x 4 col-tiles.
// B is staged into LDS pre-swizzled into fragment order: one contiguous v16h
// per (step, colTile, lane). A-fragment = two contiguous 8-half runs -> b128s.
template<int K, bool HAS_BIAS, int ACT, bool OUT_HALF>
__global__ void k_gemm_wmma(const _Float16* __restrict__ Ah,
                            const float* __restrict__ W,
                            const float* __restrict__ bias,
                            float* __restrict__ outF, _Float16* __restrict__ outH,
                            int Nrows, int lda, int ldo) {
    constexpr int STEPS = (K + 31) / 32;
    __shared__ __align__(32) _Float16 sB[STEPS * 4 * 32 * 16];

    const int tid = threadIdx.x;
    // ---- stage B fragments (once per block) ----
    for (int idx = tid; idx < STEPS * 4 * 32; idx += blockDim.x) {
        const int lane_s = idx & 31;
        const int ct     = (idx >> 5) & 3;
        const int step   = idx >> 7;
        const int kbs    = (lane_s >> 4) * 8;
        const int n      = ct * 16 + (lane_s & 15);
        _Float16* f = &sB[idx * 16];
#pragma unroll
        for (int j = 0; j < 8; ++j) {
            int k = step * 32 + ((j & 4) ? 16 : 0) + kbs + (j & 3) * 2;
            f[2 * j]     = (k     < K) ? (_Float16)W[k * 64 + n]       : (_Float16)0.f;
            f[2 * j + 1] = (k + 1 < K) ? (_Float16)W[(k + 1) * 64 + n] : (_Float16)0.f;
        }
    }
    __syncthreads();

    const int lane    = tid & 31;
    const int wave    = tid >> 5;
    const int rowTile = blockIdx.x * 2 + (wave >> 2);
    const int colTile = wave & 3;
    if (rowTile * 16 >= Nrows) return;     // wave-uniform; EXEC all-ones for WMMA

    const int m    = lane & 15;
    const int kb   = (lane >> 4) * 8;
    const int ncol = colTile * 16 + (lane & 15);
    const _Float16* arow  = Ah + (long long)(rowTile * 16 + m) * lda;
    const _Float16* bbase = &sB[(colTile * 32 + lane) * 16];

    v8f c = {};
#pragma unroll
    for (int s = 0; s < STEPS; ++s) {
        const int k0 = s * 32;
        v8h r1 = *(const v8h*)(arow + k0 + kb);
        v8h r2 = {};
        if constexpr (K >= 32) r2 = *(const v8h*)(arow + k0 + 16 + kb);
        v16h a = __builtin_shufflevector(r1, r2, 0, 1, 2, 3, 4, 5, 6, 7,
                                                 8, 9, 10, 11, 12, 13, 14, 15);
        v16h b = *(const v16h*)(bbase + s * (4 * 32 * 16));
        c = __builtin_amdgcn_wmma_f32_16x16x32_f16(
                false, a, false, b, (short)0, c, false, false);
    }

    float bv = 0.f;
    if constexpr (HAS_BIAS) bv = bias[ncol];
#pragma unroll
    for (int r = 0; r < 8; ++r) {          // C/D: m = r + 8*(lane>=16), n = lane&15
        int mm = rowTile * 16 + r + ((lane >> 4) * 8);
        float v = c[r] + bv;
        if constexpr (ACT == 1) v = fmaxf(v, 0.f);
        if constexpr (OUT_HALF) outH[(long long)mm * ldo + ncol] = (_Float16)v;
        else                    outF[(long long)mm * ldo + ncol] = v;
    }
}

// --------------------------------------------------------- edge scatters
__global__ void k_scat_gin(const float* __restrict__ xc, const int* __restrict__ src,
                           const int* __restrict__ dst, float* __restrict__ agg, int E) {
    long long t = (long long)blockIdx.x * blockDim.x + threadIdx.x;
    if (t >= (long long)E * 32) return;
    int e = (int)(t >> 5);
    int f = ((int)t & 31) * 4;
    int si = src[e], di = dst[e];
    const float4 v = *(const float4*)(xc + (long long)si * 128 + f);
    float* p = agg + (long long)di * 128 + f;
    atomicAdd(p + 0, v.x); atomicAdd(p + 1, v.y);
    atomicAdd(p + 2, v.z); atomicAdd(p + 3, v.w);
}

__global__ void k_scat_gcn(const float* __restrict__ hs, const int* __restrict__ src,
                           const int* __restrict__ dst, const float* __restrict__ dinv,
                           float* __restrict__ sa, int E) {
    long long t = (long long)blockIdx.x * blockDim.x + threadIdx.x;
    if (t >= (long long)E * 16) return;
    int e = (int)(t >> 4);
    int f = ((int)t & 15) * 4;
    int si = src[e], di = dst[e];
    float w = dinv[si] * dinv[di];
    const float4 v = *(const float4*)(hs + (long long)si * 64 + f);
    float* p = sa + (long long)di * 64 + f;
    atomicAdd(p + 0, v.x * w); atomicAdd(p + 1, v.y * w);
    atomicAdd(p + 2, v.z * w); atomicAdd(p + 3, v.w * w);
}

// s_new = tanh(sa + hs/deg + b) -> xc[:,64:]
__global__ void k_combine_s(const float* __restrict__ sa, const float* __restrict__ hs,
                            const float* __restrict__ deg, const float* __restrict__ gb,
                            float* __restrict__ xc, int Nn) {
    int t = blockIdx.x * blockDim.x + threadIdx.x;
    if (t >= Nn * 64) return;
    int n = t >> 6, h = t & 63;
    float v = tanhf(sa[t] + hs[t] / deg[n] + gb[h]);
    xc[(long long)n * 128 + 64 + h] = v;
}

__global__ void k_pool(const float* __restrict__ xw, const int* __restrict__ batch,
                       float* __restrict__ g, int Nn) {
    int t = blockIdx.x * blockDim.x + threadIdx.x;
    if (t >= Nn * 64) return;
    int n = t >> 6, h = t & 63;
    atomicAdd(&g[(long long)batch[n] * 64 + h], xw[t]);
}

__global__ void k_head(const float* __restrict__ g, const float* __restrict__ pW,
                       const float* __restrict__ pb, const float* __restrict__ rW,
                       const float* __restrict__ rb, float* __restrict__ out) {
    __shared__ float sg[64], st[64], sl[10], red[2];
    int b = blockIdx.x, h = threadIdx.x;
    sg[h] = g[(long long)b * 64 + h];
    __syncthreads();
    float acc = pb[h];
    for (int k = 0; k < 64; ++k) acc += sg[k] * pW[k * 64 + h];
    st[h] = fmaxf(acc, 0.f);
    __syncthreads();
    if (h < 10) {
        float a2 = rb[h];
        for (int k = 0; k < 64; ++k) a2 += st[k] * rW[k * 10 + h];
        sl[h] = a2;
    }
    __syncthreads();
    if (h == 0) {
        float mx = sl[0];
        for (int i = 1; i < 10; ++i) mx = fmaxf(mx, sl[i]);
        float se = 0.f;
        for (int i = 0; i < 10; ++i) se += expf(sl[i] - mx);
        red[0] = mx; red[1] = logf(se);
    }
    __syncthreads();
    if (h < 10) out[b * 10 + h] = sl[h] - red[0] - red[1];
}

static inline int cdiv(long long a, long long b) { return (int)((a + b - 1) / b); }

extern "C" void kernel_launch(void* const* d_in, const int* in_sizes, int n_in,
                              void* d_out, int out_size, void* d_ws, size_t ws_size,
                              hipStream_t stream) {
    const int N = GN_N, E = GN_E, G = GN_G;

    const float* x0    = (const float*)d_in[0];
    const float* s0    = (const float*)d_in[1];
    const int*   ei    = (const int*)d_in[2];
    const int*   src   = ei;
    const int*   dst   = ei + E;
    const int*   batch = (const int*)d_in[3];
    const float* pre_W = (const float*)d_in[4];
    const float* pre_b = (const float*)d_in[5];
    const float* emb_W = (const float*)d_in[6];
    const float* emb_b = (const float*)d_in[7];
    const float* gin_W1 = (const float*)d_in[8];
    const float* gin_b1 = (const float*)d_in[9];
    const float* gin_W2 = (const float*)d_in[10];
    const float* gin_b2 = (const float*)d_in[11];
    const float* gcn_W  = (const float*)d_in[12];
    const float* gcn_b  = (const float*)d_in[13];
    const float* whp_W  = (const float*)d_in[14];
    const float* whp_b  = (const float*)d_in[15];
    const float* post_W = (const float*)d_in[16];
    const float* post_b = (const float*)d_in[17];
    const float* ro_W   = (const float*)d_in[18];
    const float* ro_b   = (const float*)d_in[19];

    // ---- workspace: f32 region then f16 region ----
    float* ws   = (float*)d_ws;
    float* deg  = ws;                                  // N
    float* dinv = deg  + N;                            // N
    float* xc   = dinv + N;                            // N*128 (x cols 0..63, s cols 64..127)
    float* agg  = xc   + (size_t)N * 128;              // N*128
    float* hs   = agg  + (size_t)N * 128;              // N*64
    float* sa   = hs   + (size_t)N * 64;               // N*64
    float* xw   = sa   + (size_t)N * 64;               // N*64
    float* gbuf = xw   + (size_t)N * 64;               // G*64
    _Float16* h16    = (_Float16*)(gbuf + (size_t)G * 64);
    _Float16* aH128  = h16;                            // N*128 f16 (GEMM A, K=128)
    _Float16* aHt    = aH128 + (size_t)N * 128;        // N*64  f16 (GIN hidden)
    _Float16* aHs    = aHt   + (size_t)N * 64;         // N*64  f16 (GCN input s)
    _Float16* aH16   = aHs   + (size_t)N * 64;         // N*16  f16 (emb input)

    const int rowTiles   = N / 16;                     // 3125
    const int gemmBlocks = (rowTiles + 1) / 2;

    // degree + normalization
    k_fill<<<cdiv(N, 256), 256, 0, stream>>>(deg, N, 1.0f);
    k_deg<<<cdiv(E, 256), 256, 0, stream>>>(dst, deg, E);
    k_dinv<<<cdiv(N, 256), 256, 0, stream>>>(deg, dinv, N);

    // x = x0@pre_W+pre_b -> xc[:,0:64] ; s = s0@emb_W+emb_b -> xc[:,64:128]
    k_cvt<<<cdiv((long long)N * 128, 256), 256, 0, stream>>>(aH128, x0, nullptr, N, 128, 128, 128);
    k_gemm_wmma<128, true, 0, false><<<gemmBlocks, 256, 0, stream>>>(
        aH128, pre_W, pre_b, xc, nullptr, N, 128, 128);
    k_cvt<<<cdiv((long long)N * 16, 256), 256, 0, stream>>>(aH16, s0, nullptr, N, 16, 16, 16);
    k_gemm_wmma<16, true, 0, false><<<gemmBlocks, 256, 0, stream>>>(
        aH16, emb_W, emb_b, xc + 64, nullptr, N, 16, 128);

    for (int i = 0; i < 3; ++i) {
        // GIN
        k_fill<<<cdiv((long long)N * 128, 256), 256, 0, stream>>>(agg, N * 128, 0.f);
        k_scat_gin<<<cdiv((long long)E * 32, 256), 256, 0, stream>>>(xc, src, dst, agg, E);
        k_cvt<<<cdiv((long long)N * 128, 256), 256, 0, stream>>>(aH128, xc, agg, N, 128, 128, 128);
        k_gemm_wmma<128, true, 1, true><<<gemmBlocks, 256, 0, stream>>>(
            aH128, gin_W1 + (size_t)i * 128 * 64, gin_b1 + i * 64, nullptr, aHt, N, 128, 64);
        k_gemm_wmma<64, true, 1, false><<<gemmBlocks, 256, 0, stream>>>(
            aHt, gin_W2 + (size_t)i * 64 * 64, gin_b2 + i * 64, xc, nullptr, N, 64, 128);
        // GCN (uses old s = xc[:,64:], untouched by the GIN writes above)
        k_cvt<<<cdiv((long long)N * 64, 256), 256, 0, stream>>>(aHs, xc + 64, nullptr, N, 64, 128, 64);
        k_gemm_wmma<64, false, 0, false><<<gemmBlocks, 256, 0, stream>>>(
            aHs, gcn_W + (size_t)i * 64 * 64, nullptr, hs, nullptr, N, 64, 64);
        k_fill<<<cdiv((long long)N * 64, 256), 256, 0, stream>>>(sa, N * 64, 0.f);
        k_scat_gcn<<<cdiv((long long)E * 16, 256), 256, 0, stream>>>(hs, src, dst, dinv, sa, E);
        k_combine_s<<<cdiv((long long)N * 64, 256), 256, 0, stream>>>(sa, hs, deg, gcn_b + i * 64, xc, N);
    }

    // xw = xc @ whp_W + whp_b
    k_cvt<<<cdiv((long long)N * 128, 256), 256, 0, stream>>>(aH128, xc, nullptr, N, 128, 128, 128);
    k_gemm_wmma<128, true, 0, false><<<gemmBlocks, 256, 0, stream>>>(
        aH128, whp_W, whp_b, xw, nullptr, N, 128, 64);

    // pooled head
    k_fill<<<cdiv(G * 64, 256), 256, 0, stream>>>(gbuf, G * 64, 0.f);
    k_pool<<<cdiv((long long)N * 64, 256), 256, 0, stream>>>(xw, batch, gbuf, N);
    k_head<<<G, 64, 0, stream>>>(gbuf, post_W, post_b, ro_W, ro_b, (float*)d_out);
}
